// Kandinsky5TransformerEncoderBlock_68032281968713
// MI455X (gfx1250) — matrix operand order repaired and checked by reading, not compiled, not run on hardware
//
#include <hip/hip_runtime.h>

// ---------------------------------------------------------------------------
// Kandinsky5 transformer encoder block on MI455X (gfx1250, wave32, WMMA).
// f16 WMMA (v_wmma_f32_16x16x32_f16) with fp32 accumulate on the matrix path;
// Tensor Data Mover (tensor_load_to_lds + s_wait_tensorcnt) double-buffered
// LDS staging; fp32 VALU for norms/softmax/rope/modulation.
// ---------------------------------------------------------------------------

#define B_  1
#define S_  4096
#define MD_ 1024
#define TD_ 512
#define FF_ 4096
#define HD_ 128
#define H_  8

typedef __attribute__((ext_vector_type(16))) _Float16 v16h;
typedef __attribute__((ext_vector_type(8)))  float    v8f;
typedef __attribute__((ext_vector_type(4)))  unsigned u32x4;
typedef __attribute__((ext_vector_type(8)))  int      i32x8;
typedef __attribute__((ext_vector_type(4)))  int      i32x4;

union H2 { unsigned u; _Float16 h[2]; };

#if __has_builtin(__builtin_amdgcn_tensor_load_to_lds) && \
    __has_builtin(__builtin_amdgcn_s_wait_tensorcnt)
#define HAVE_TDM 1
#else
#define HAVE_TDM 0
#endif

__device__ __forceinline__ v8f wmma32(v16h a, v16h b, v8f c) {
  return __builtin_amdgcn_wmma_f32_16x16x32_f16(false, a, false, b, (short)0, c, false, false);
}

// Low 32 bits of a generic pointer to LDS = LDS byte offset (aperture layout).
__device__ __forceinline__ unsigned lds_off(const void* p) {
  return (unsigned)(unsigned long long)p;
}

#if HAVE_TDM
// Issue a TDM 2D-tile load: tile (tile_d1 rows x tile_d0 elems) of 2-byte
// elements from a row-major tensor (row stride stride_e elements) into LDS,
// packed contiguously (tile_d0*2 bytes per row).  D# bitfields per ISA §8.3-8.4.
// This toolchain's builtin is the 6-arg form:
//   (u32x4 g0, i32x8 g1, i32x4 g2, i32x4 g3, i32x8 g4, i32 cpol)
__device__ __forceinline__ void tdm_load_2d(
    unsigned lds_byte, const void* gaddr, unsigned tensor_d0,
    unsigned tensor_d1, unsigned tile_d0, unsigned tile_d1,
    unsigned long long stride_e) {
  const unsigned long long ga = (unsigned long long)gaddr;
  u32x4 g0;
  g0[0] = 1u;                                              // count=1 (valid D#)
  g0[1] = lds_byte;                                        // lds_addr
  g0[2] = (unsigned)(ga & 0xffffffffull);                  // global_addr[31:0]
  g0[3] = (unsigned)((ga >> 32) & 0x01ffffffull) | (2u << 30);  // [56:32]|type=2
  i32x8 g1;
  g1[0] = (int)(1u << 16);                                 // data_size=1 (2B)
  g1[1] = (int)((tensor_d0 & 0xffffu) << 16);              // tensor_dim0[15:0]
  g1[2] = (int)(((tensor_d0 >> 16) & 0xffffu) | ((tensor_d1 & 0xffffu) << 16));
  g1[3] = (int)(((tensor_d1 >> 16) & 0xffffu) | ((tile_d0 & 0xffffu) << 16));
  g1[4] = (int)(tile_d1 & 0xffffu);                        // tile_dim1, dim2=0
  g1[5] = (int)(unsigned)(stride_e & 0xffffffffull);       // dim0_stride[31:0]
  g1[6] = (int)(unsigned)((stride_e >> 32) & 0xffffull);   // dim0_stride[47:32]
  g1[7] = 0;
  const i32x4 gz4 = {0, 0, 0, 0};
  const i32x8 gz8 = {0, 0, 0, 0, 0, 0, 0, 0};
  __builtin_amdgcn_tensor_load_to_lds(g0, g1, gz4, gz4, gz8, 0);
}
#endif

// A-matrix fragment, 16x32 f16 (M x K), memory row-major base[row*ld + k].
__device__ __forceinline__ v16h load_frag_a(const _Float16* base, int ld) {
  const int lane = threadIdx.x & 31;
  const int row  = lane & 15;
  const int hi   = (lane >> 4) & 1;
  v16h a;
#pragma unroll
  for (int i = 0; i < 8; ++i) {
    const int k = ((i >> 2) << 4) + (hi << 3) + ((i & 3) << 1);
    H2 u; u.u = *(const unsigned*)(base + row * ld + k);
    a[2 * i] = u.h[0]; a[2 * i + 1] = u.h[1];
  }
  return a;
}

// B-matrix fragment, 32x16 f16 (K x N), with B[k][n] stored at base[n*ld + k].
__device__ __forceinline__ v16h load_frag_b(const _Float16* base, int ld) {
  const int lane = threadIdx.x & 31;
  const int n    = lane & 15;
  const int hi   = (lane >> 4) & 1;
  v16h b;
#pragma unroll
  for (int i = 0; i < 8; ++i) {
    const int k = ((i >> 2) << 4) + (hi << 3) + ((i & 3) << 1);
    H2 u; u.u = *(const unsigned*)(base + n * ld + k);
    b[2 * i] = u.h[0]; b[2 * i + 1] = u.h[1];
  }
  return b;
}

// ---------------------------------------------------------------------------
// 1) modulation: mod[o] = sum_t silu(te[t]) * W_mod[o,t] + b_mod[o]   (6144)
// ---------------------------------------------------------------------------
__global__ __launch_bounds__(256) void mod_kernel(
    const float* __restrict__ te, const float* __restrict__ Wm,
    const float* __restrict__ bm, float* __restrict__ modv) {
  const int o = blockIdx.x * 256 + threadIdx.x;
  float acc = 0.f;
  for (int t = 0; t < TD_; ++t) {
    const float v  = te[t];
    const float sl = v / (1.f + __expf(-v));
    acc += sl * Wm[(size_t)o * TD_ + t];
  }
  modv[o] = acc + bm[o];
}

// ---------------------------------------------------------------------------
// 2) fp32 -> f16 weight conversion (vectorized x4)
// ---------------------------------------------------------------------------
__global__ __launch_bounds__(256) void cvt_f16(
    const float* __restrict__ in, _Float16* __restrict__ out, int n4) {
  const int i = blockIdx.x * 256 + threadIdx.x;
  if (i < n4) {
    const float4 v = ((const float4*)in)[i];
    H2 a, b;
    a.h[0] = (_Float16)v.x; a.h[1] = (_Float16)v.y;
    b.h[0] = (_Float16)v.z; b.h[1] = (_Float16)v.w;
    ((unsigned*)out)[2 * i + 0] = a.u;
    ((unsigned*)out)[2 * i + 1] = b.u;
  }
}

// ---------------------------------------------------------------------------
// 3) LayerNorm (no affine) + AdaLN modulate -> f16.  One block per row.
// ---------------------------------------------------------------------------
__global__ __launch_bounds__(256) void ln_mod(
    const float* __restrict__ x, const float* __restrict__ modv, int modoff,
    _Float16* __restrict__ hout) {
  __shared__ float red[16];
  const int s = blockIdx.x, t = threadIdx.x;
  const int lane = t & 31, wid = t >> 5;
  union { float4 v; float a[4]; } xv;
  xv.v = *(const float4*)(x + (size_t)s * MD_ + t * 4);
  float s1 = xv.a[0] + xv.a[1] + xv.a[2] + xv.a[3];
  float s2 = xv.a[0]*xv.a[0] + xv.a[1]*xv.a[1] + xv.a[2]*xv.a[2] + xv.a[3]*xv.a[3];
#pragma unroll
  for (int off = 16; off; off >>= 1) {
    s1 += __shfl_xor(s1, off, 32);
    s2 += __shfl_xor(s2, off, 32);
  }
  if (lane == 0) { red[wid] = s1; red[8 + wid] = s2; }
  __syncthreads();
  float S1 = 0.f, S2 = 0.f;
#pragma unroll
  for (int w = 0; w < 8; ++w) { S1 += red[w]; S2 += red[8 + w]; }
  const float mean = S1 * (1.f / MD_);
  const float var  = S2 * (1.f / MD_) - mean * mean;
  const float rinv = rsqrtf(var + 1e-5f);
#pragma unroll
  for (int e = 0; e < 4; ++e) {
    const int c = t * 4 + e;
    const float sc = modv[modoff + MD_ + c] + 1.f;
    const float sh = modv[modoff + c];
    hout[(size_t)s * MD_ + c] = (_Float16)((xv.a[e] - mean) * rinv * sc + sh);
  }
}

// ---------------------------------------------------------------------------
// 4) WMMA GEMM: C[M,N] = A[M,K](f16,rm) * B[N,K](f16,rm)^T  (+epilogue).
//    Block tile 64x128, 8 waves each 32x32, BK=32.  TDM double-buffered LDS:
//    wave0 DMAs the A tile, wave1 the B tile, s_wait_tensorcnt pipelines.
//    EPI 0: outF = acc + bias[col]
//    EPI 1: outH = f16(gelu_exact(acc))
//    EPI 2: outF = res + gate[col]*(acc + bias[col])
//    EPI 3: outF = res + gate[col]*acc
// ---------------------------------------------------------------------------
template <int EPI>
__global__ __launch_bounds__(256) void gemm_f16(
    const _Float16* __restrict__ A, const _Float16* __restrict__ Bm,
    const float* __restrict__ bias, const float* __restrict__ res,
    const float* __restrict__ gate, float* __restrict__ outF,
    _Float16* __restrict__ outH, int M, int N, int K) {
  __shared__ __align__(16) _Float16 sA[2][64 * 32];
  __shared__ __align__(16) _Float16 sB[2][128 * 32];
  const int t = threadIdx.x;
  const int lane = t & 31, wid = t >> 5;
  const int wm = wid >> 2, wn = wid & 3;   // 2 x 4 waves
  const int m0 = blockIdx.y * 64, n0 = blockIdx.x * 128;
  const int hi = (lane >> 4) & 1, nn = lane & 15;

  v8f acc[2][2] = {};
  const int nk = K / 32;

#if HAVE_TDM
  if (wid == 0)
    tdm_load_2d(lds_off(&sA[0][0]), A + (size_t)m0 * K, K, M, 32, 64, K);
  else if (wid == 1)
    tdm_load_2d(lds_off(&sB[0][0]), Bm + (size_t)n0 * K, K, N, 32, 128, K);
#endif

  for (int kt = 0; kt < nk; ++kt) {
    const int cur = kt & 1;
#if HAVE_TDM
    if (kt + 1 < nk) {
      const int k1 = (kt + 1) * 32;
      if (wid == 0)
        tdm_load_2d(lds_off(&sA[cur ^ 1][0]), A + (size_t)m0 * K + k1,
                    K, M, 32, 64, K);
      else if (wid == 1)
        tdm_load_2d(lds_off(&sB[cur ^ 1][0]), Bm + (size_t)n0 * K + k1,
                    K, N, 32, 128, K);
      if (wid < 2) __builtin_amdgcn_s_wait_tensorcnt(1);
    } else if (wid < 2) {
      __builtin_amdgcn_s_wait_tensorcnt(0);
    }
#else
    {  // fallback: cooperative vector loads
      const int k0 = kt * 32;
      const int r = t >> 2, c = (t & 3) << 3;
      *(float4*)(sA[cur] + r * 32 + c) =
          *(const float4*)(A + (size_t)(m0 + r) * K + k0 + c);
#pragma unroll
      for (int it2 = 0; it2 < 2; ++it2) {
        const int idx = t + it2 * 256;
        const int rb = idx >> 2, cb = (idx & 3) << 3;
        *(float4*)(sB[cur] + rb * 32 + cb) =
            *(const float4*)(Bm + (size_t)(n0 + rb) * K + k0 + cb);
      }
    }
#endif
    __syncthreads();
    const _Float16* cA = sA[cur];
    const _Float16* cB = sB[cur];
    const v16h a0 = load_frag_a(cA + (wm * 32 +  0) * 32, 32);
    const v16h a1 = load_frag_a(cA + (wm * 32 + 16) * 32, 32);
    const v16h b0 = load_frag_b(cB + (wn * 32 +  0) * 32, 32);
    const v16h b1 = load_frag_b(cB + (wn * 32 + 16) * 32, 32);
    acc[0][0] = wmma32(a0, b0, acc[0][0]);
    acc[0][1] = wmma32(a0, b1, acc[0][1]);
    acc[1][0] = wmma32(a1, b0, acc[1][0]);
    acc[1][1] = wmma32(a1, b1, acc[1][1]);
    __syncthreads();
  }

#pragma unroll
  for (int i = 0; i < 2; ++i)
#pragma unroll
    for (int j = 0; j < 2; ++j)
#pragma unroll
      for (int r = 0; r < 8; ++r) {
        const int row = m0 + wm * 32 + i * 16 + hi * 8 + r;
        const int col = n0 + wn * 32 + j * 16 + nn;
        const size_t idx = (size_t)row * N + col;
        const float v = acc[i][j][r];
        if (EPI == 0) {
          outF[idx] = v + bias[col];
        } else if (EPI == 1) {
          const float g = 0.5f * v * (1.f + erff(v * 0.70710678118654752f));
          outH[idx] = (_Float16)g;
        } else if (EPI == 2) {
          outF[idx] = res[idx] + gate[col] * (v + bias[col]);
        } else {
          outF[idx] = res[idx] + gate[col] * v;
        }
      }
}

// ---------------------------------------------------------------------------
// 5) per-head RMSNorm + RoPE + 1/sqrt(HD) fold (q only); fp32 -> f16.
//    q,k head-major [H,S,HD]; v transposed d-major [H,HD,S] so attention's
//    P*V B-fragments are contiguous-K LDS loads.
// ---------------------------------------------------------------------------
__global__ __launch_bounds__(256) void qkv_norm_rope(
    const float* __restrict__ Q32, const float* __restrict__ K32,
    const float* __restrict__ V32, const float* __restrict__ rope,
    const float* __restrict__ qw, const float* __restrict__ kw,
    _Float16* __restrict__ qf, _Float16* __restrict__ kf,
    _Float16* __restrict__ vt) {
  const int s = blockIdx.x;
  const int lane = threadIdx.x & 31, h = threadIdx.x >> 5;
  const size_t src = (size_t)s * MD_ + h * HD_ + lane * 4;
  union { float4 v; float a[4]; } q4, k4, v4;
  q4.v = *(const float4*)(Q32 + src);
  k4.v = *(const float4*)(K32 + src);
  v4.v = *(const float4*)(V32 + src);

  float qs = q4.a[0]*q4.a[0] + q4.a[1]*q4.a[1] + q4.a[2]*q4.a[2] + q4.a[3]*q4.a[3];
  float ks = k4.a[0]*k4.a[0] + k4.a[1]*k4.a[1] + k4.a[2]*k4.a[2] + k4.a[3]*k4.a[3];
#pragma unroll
  for (int off = 16; off; off >>= 1) {
    qs += __shfl_xor(qs, off, 32);
    ks += __shfl_xor(ks, off, 32);
  }
  const float EPSR = 1.1920928955078125e-07f;  // fp32 eps (RMSNorm default)
  const float qr = rsqrtf(qs * (1.f / HD_) + EPSR);
  const float kr = rsqrtf(ks * (1.f / HD_) + EPSR);

  float qn[4], kn[4];
#pragma unroll
  for (int e = 0; e < 4; ++e) {
    qn[e] = q4.a[e] * qr * qw[lane * 4 + e];
    kn[e] = k4.a[e] * kr * kw[lane * 4 + e];
  }
  // rope[(s*64 + i)*4 + j*2 + r]; this lane owns pairs i = lane*2, lane*2+1
  const float* rp = rope + (size_t)s * 256 + lane * 8;
  const float SCALE = 0.08838834764831845f;  // 1/sqrt(128) folded into q
  float qo[4], ko[4];
  qo[0] = (rp[0] * qn[0] + rp[1] * qn[1]) * SCALE;
  qo[1] = (rp[2] * qn[0] + rp[3] * qn[1]) * SCALE;
  qo[2] = (rp[4] * qn[2] + rp[5] * qn[3]) * SCALE;
  qo[3] = (rp[6] * qn[2] + rp[7] * qn[3]) * SCALE;
  ko[0] = rp[0] * kn[0] + rp[1] * kn[1];
  ko[1] = rp[2] * kn[0] + rp[3] * kn[1];
  ko[2] = rp[4] * kn[2] + rp[5] * kn[3];
  ko[3] = rp[6] * kn[2] + rp[7] * kn[3];

  const size_t dst = ((size_t)h * S_ + s) * HD_ + lane * 4;
  H2 p0, p1;
  p0.h[0] = (_Float16)qo[0]; p0.h[1] = (_Float16)qo[1];
  p1.h[0] = (_Float16)qo[2]; p1.h[1] = (_Float16)qo[3];
  *(unsigned*)(qf + dst) = p0.u; *(unsigned*)(qf + dst + 2) = p1.u;
  p0.h[0] = (_Float16)ko[0]; p0.h[1] = (_Float16)ko[1];
  p1.h[0] = (_Float16)ko[2]; p1.h[1] = (_Float16)ko[3];
  *(unsigned*)(kf + dst) = p0.u; *(unsigned*)(kf + dst + 2) = p1.u;
  const size_t vdst = ((size_t)h * HD_ + lane * 4) * S_ + s;  // d-major
#pragma unroll
  for (int e = 0; e < 4; ++e) vt[vdst + (size_t)e * S_] = (_Float16)v4.a[e];
}

// ---------------------------------------------------------------------------
// 6) Flash attention.  grid = (S/64, H), 128 threads (4 waves).
//    Each wave owns 16 query rows; 64-key K (row-major) and V (d-major)
//    tiles DMA'd into double-buffered LDS by waves 0/1 via TDM.
// ---------------------------------------------------------------------------
__global__ __launch_bounds__(128) void attn_kernel(
    const _Float16* __restrict__ qf, const _Float16* __restrict__ kf,
    const _Float16* __restrict__ vt, _Float16* __restrict__ attn16) {
  __shared__ __align__(16) _Float16 sK[2][64 * 128];   // [key][d]
  __shared__ __align__(16) _Float16 sVt[2][128 * 64];  // [d][key]
  __shared__ __align__(16) _Float16 sP[4][16 * 64];
  const int t = threadIdx.x, lane = t & 31, wid = t >> 5;
  const int h = blockIdx.y;
  const int q0 = blockIdx.x * 64 + wid * 16;
  const int hi = (lane >> 4) & 1, nn = lane & 15;

  const _Float16* qbase = qf + ((size_t)h * S_ + q0) * HD_;
  v16h aQ[4];
#pragma unroll
  for (int ka = 0; ka < 4; ++ka) aQ[ka] = load_frag_a(qbase + ka * 32, HD_);

  v8f o[8] = {};
  float mrow[8], lrow[8];
#pragma unroll
  for (int r = 0; r < 8; ++r) { mrow[r] = -3.0e30f; lrow[r] = 0.f; }

  const int nk = S_ / 64;
#if HAVE_TDM
  if (wid == 0)
    tdm_load_2d(lds_off(&sK[0][0]), kf + (size_t)h * S_ * HD_,
                HD_, S_, HD_, 64, HD_);
  else if (wid == 1)
    tdm_load_2d(lds_off(&sVt[0][0]), vt + (size_t)h * HD_ * S_,
                S_, HD_, 64, HD_, S_);
#endif

  for (int it = 0; it < nk; ++it) {
    const int cur = it & 1;
#if HAVE_TDM
    if (it + 1 < nk) {
      const int kt1 = (it + 1) * 64;
      if (wid == 0)
        tdm_load_2d(lds_off(&sK[cur ^ 1][0]),
                    kf + ((size_t)h * S_ + kt1) * HD_, HD_, S_, HD_, 64, HD_);
      else if (wid == 1)
        tdm_load_2d(lds_off(&sVt[cur ^ 1][0]),
                    vt + (size_t)h * HD_ * S_ + kt1, S_, HD_, 64, HD_, S_);
      if (wid < 2) __builtin_amdgcn_s_wait_tensorcnt(1);
    } else if (wid < 2) {
      __builtin_amdgcn_s_wait_tensorcnt(0);
    }
#else
    {
      const int kt = it * 64;
#pragma unroll
      for (int i2 = 0; i2 < 4; ++i2) {
        const int idx = t + i2 * 128;
        const int r = idx >> 3, c = (idx & 7) << 4;
        *(float4*)(sK[cur] + r * 128 + c) =
            *(const float4*)(kf + ((size_t)h * S_ + kt + r) * HD_ + c);
      }
#pragma unroll
      for (int i2 = 0; i2 < 4; ++i2) {
        const int idx = t + i2 * 128;
        const int r = idx >> 2, c = (idx & 3) << 4;  // 128 rows x 64 cols
        *(float4*)(sVt[cur] + r * 64 + c) =
            *(const float4*)(vt + ((size_t)h * HD_ + r) * S_ + kt + c);
      }
    }
#endif
    __syncthreads();

    // S tile = Q(16x128) * K_tile^T -> 16x64 fp32
    v8f sacc[4] = {};
#pragma unroll
    for (int jn = 0; jn < 4; ++jn)
#pragma unroll
      for (int ka = 0; ka < 4; ++ka) {
        const v16h b = load_frag_b(sK[cur] + (jn * 16) * 128 + ka * 32, 128);
        sacc[jn] = wmma32(aQ[ka], b, sacc[jn]);
      }

    // online softmax (row = r + 8*hi; 16 lanes per row-half share a row)
#pragma unroll
    for (int r = 0; r < 8; ++r) {
      float mx = fmaxf(fmaxf(sacc[0][r], sacc[1][r]),
                       fmaxf(sacc[2][r], sacc[3][r]));
#pragma unroll
      for (int off = 8; off; off >>= 1) mx = fmaxf(mx, __shfl_xor(mx, off, 16));
      const float mnew  = fmaxf(mrow[r], mx);
      const float alpha = __expf(mrow[r] - mnew);
      float ssum = 0.f;
#pragma unroll
      for (int jn = 0; jn < 4; ++jn) {
        const float p = __expf(sacc[jn][r] - mnew);
        ssum += p;
        sP[wid][(r + hi * 8) * 64 + jn * 16 + nn] = (_Float16)p;
      }
#pragma unroll
      for (int off = 8; off; off >>= 1) ssum += __shfl_xor(ssum, off, 16);
      lrow[r] = lrow[r] * alpha + ssum;
      mrow[r] = mnew;
#pragma unroll
      for (int jn2 = 0; jn2 < 8; ++jn2) o[jn2][r] *= alpha;
    }

    // O += P(16x64) * V_tile(64x128); V is d-major so B-frags are K-contiguous
    const v16h aP0 = load_frag_a(&sP[wid][0], 64);
    const v16h aP1 = load_frag_a(&sP[wid][32], 64);
#pragma unroll
    for (int jn2 = 0; jn2 < 8; ++jn2) {
      const v16h bv0 = load_frag_b(sVt[cur] + (jn2 * 16) * 64 +  0, 64);
      const v16h bv1 = load_frag_b(sVt[cur] + (jn2 * 16) * 64 + 32, 64);
      o[jn2] = wmma32(aP0, bv0, o[jn2]);
      o[jn2] = wmma32(aP1, bv1, o[jn2]);
    }
    __syncthreads();
  }

#pragma unroll
  for (int jn2 = 0; jn2 < 8; ++jn2)
#pragma unroll
    for (int r = 0; r < 8; ++r) {
      const int srow = q0 + r + hi * 8;
      const float val = o[jn2][r] / lrow[r];
      attn16[(size_t)srow * MD_ + h * HD_ + jn2 * 16 + nn] = (_Float16)val;
    }
}

// ---------------------------------------------------------------------------
// host-side orchestration
// ---------------------------------------------------------------------------
extern "C" void kernel_launch(void* const* d_in, const int* in_sizes, int n_in,
                              void* d_out, int out_size, void* d_ws,
                              size_t ws_size, hipStream_t stream) {
  const float* x    = (const float*)d_in[0];
  const float* te   = (const float*)d_in[1];
  const float* rope = (const float*)d_in[2];
  const float* Wmod = (const float*)d_in[3];
  const float* bmod = (const float*)d_in[4];
  const float* Wq   = (const float*)d_in[5];
  const float* bq   = (const float*)d_in[6];
  const float* Wk   = (const float*)d_in[7];
  const float* bk   = (const float*)d_in[8];
  const float* Wv   = (const float*)d_in[9];
  const float* bv   = (const float*)d_in[10];
  const float* qw   = (const float*)d_in[11];
  const float* kw   = (const float*)d_in[12];
  const float* Wo   = (const float*)d_in[13];
  const float* bo   = (const float*)d_in[14];
  const float* Wfi  = (const float*)d_in[15];
  const float* Wfo  = (const float*)d_in[16];
  float* out = (float*)d_out;

  char* ws = (char*)d_ws;
  size_t off = 0;
  auto take = [&](size_t b) {
    size_t o = off;
    off += (b + 255) & ~(size_t)255;
    return o;
  };
  float*    modv = (float*)(ws + take(6144 * sizeof(float)));
  _Float16* wq16 = (_Float16*)(ws + take((size_t)MD_ * MD_ * 2));
  _Float16* wk16 = (_Float16*)(ws + take((size_t)MD_ * MD_ * 2));
  _Float16* wv16 = (_Float16*)(ws + take((size_t)MD_ * MD_ * 2));
  _Float16* wo16 = (_Float16*)(ws + take((size_t)MD_ * MD_ * 2));
  _Float16* wfi16 = (_Float16*)(ws + take((size_t)FF_ * MD_ * 2));
  _Float16* wfo16 = (_Float16*)(ws + take((size_t)MD_ * FF_ * 2));
  _Float16* hbuf = (_Float16*)(ws + take((size_t)S_ * MD_ * 2));  // h/attn/h2
  _Float16* qf = (_Float16*)(ws + take((size_t)H_ * S_ * HD_ * 2));
  _Float16* kf = (_Float16*)(ws + take((size_t)H_ * S_ * HD_ * 2));
  _Float16* vt = (_Float16*)(ws + take((size_t)H_ * S_ * HD_ * 2));
  size_t o_q32 = take((size_t)S_ * MD_ * 4);
  size_t o_k32 = take((size_t)S_ * MD_ * 4);
  size_t o_v32 = take((size_t)S_ * MD_ * 4);
  float* q32 = (float*)(ws + o_q32);
  float* k32 = (float*)(ws + o_k32);
  float* v32 = (float*)(ws + o_v32);
  float* x1 = (float*)(ws + take((size_t)S_ * MD_ * 4));
  // GELU activations (S x FF, f16, 32MB) alias the dead q32+k32 fp32 buffers.
  _Float16* g16 = (_Float16*)(ws + o_q32);

  mod_kernel<<<6144 / 256, 256, 0, stream>>>(te, Wmod, bmod, modv);

  const int n4a = (MD_ * MD_) / 4, n4b = (FF_ * MD_) / 4;
  cvt_f16<<<(n4a + 255) / 256, 256, 0, stream>>>(Wq, wq16, n4a);
  cvt_f16<<<(n4a + 255) / 256, 256, 0, stream>>>(Wk, wk16, n4a);
  cvt_f16<<<(n4a + 255) / 256, 256, 0, stream>>>(Wv, wv16, n4a);
  cvt_f16<<<(n4a + 255) / 256, 256, 0, stream>>>(Wo, wo16, n4a);
  cvt_f16<<<(n4b + 255) / 256, 256, 0, stream>>>(Wfi, wfi16, n4b);
  cvt_f16<<<(n4b + 255) / 256, 256, 0, stream>>>(Wfo, wfo16, n4b);

  ln_mod<<<S_, 256, 0, stream>>>(x, modv, 0, hbuf);

  dim3 g1(MD_ / 128, S_ / 64);
  gemm_f16<0><<<g1, 256, 0, stream>>>(hbuf, wq16, bq, nullptr, nullptr, q32,
                                      nullptr, S_, MD_, MD_);
  gemm_f16<0><<<g1, 256, 0, stream>>>(hbuf, wk16, bk, nullptr, nullptr, k32,
                                      nullptr, S_, MD_, MD_);
  gemm_f16<0><<<g1, 256, 0, stream>>>(hbuf, wv16, bv, nullptr, nullptr, v32,
                                      nullptr, S_, MD_, MD_);

  qkv_norm_rope<<<S_, 256, 0, stream>>>(q32, k32, v32, rope, qw, kw, qf, kf, vt);

  attn_kernel<<<dim3(S_ / 64, H_), 128, 0, stream>>>(qf, kf, vt, hbuf);

  gemm_f16<2><<<g1, 256, 0, stream>>>(hbuf, wo16, bo, x, modv + 2048, x1,
                                      nullptr, S_, MD_, MD_);

  ln_mod<<<S_, 256, 0, stream>>>(x1, modv, 3072, hbuf);

  dim3 g2(FF_ / 128, S_ / 64);
  gemm_f16<1><<<g2, 256, 0, stream>>>(hbuf, wfi16, nullptr, nullptr, nullptr,
                                      nullptr, g16, S_, FF_, MD_);

  gemm_f16<3><<<g1, 256, 0, stream>>>(g16, wfo16, nullptr, x1, modv + 5120, out,
                                      nullptr, S_, MD_, FF_);
}
